// MaskedSelfAttention_56032143343859
// MI455X (gfx1250) — matrix operand order repaired and checked by reading, not compiled, not run on hardware
//
#include <hip/hip_runtime.h>

typedef __attribute__((ext_vector_type(16))) __bf16 v16bf;
typedef __attribute__((ext_vector_type(8)))  __bf16 v8bf;
typedef __attribute__((ext_vector_type(4)))  __bf16 v4bf;
typedef __attribute__((ext_vector_type(8)))  float  v8f;
typedef __attribute__((ext_vector_type(4)))  float  v4f;

#define SEQ 8192
#define DM  1024
#define DK  128
#define WPB 4           // waves per block in attention

// Combine two 8-element bf16 groups into one WMMA 16-elem operand.
__device__ __forceinline__ v16bf pack16(v8bf a, v8bf b) {
  v16bf r;
#pragma unroll
  for (int i = 0; i < 8; ++i) { r[i] = a[i]; r[i + 8] = b[i]; }
  return r;
}

// Load 8+8 bf16 from two (16B-aligned) addresses.
__device__ __forceinline__ v16bf load16bf(const __bf16* p0, const __bf16* p1) {
  return pack16(*(const v8bf*)p0, *(const v8bf*)p1);
}

// Convert 8+8 fp32 from two (16B-aligned) addresses into a bf16 operand.
__device__ __forceinline__ v16bf cvt16bf(const float* p0, const float* p1) {
  v4f a0 = *(const v4f*)p0, a1 = *(const v4f*)(p0 + 4);
  v4f b0 = *(const v4f*)p1, b1 = *(const v4f*)(p1 + 4);
  v16bf r;
#pragma unroll
  for (int i = 0; i < 4; ++i) {
    r[i]      = (__bf16)a0[i];
    r[i + 4]  = (__bf16)a1[i];
    r[i + 8]  = (__bf16)b0[i];
    r[i + 12] = (__bf16)b1[i];
  }
  return r;
}

__device__ __forceinline__ v8f wmma_bf16(v16bf a, v16bf b, v8f c) {
  return __builtin_amdgcn_wmma_f32_16x16x32_bf16(false, a, false, b,
                                                 (short)0, c, false, false);
}

// ---------------------------------------------------------------------------
// One-shot fp32 -> bf16 conversion of Wq|Wk|Wv into Wb[3][128*1024].
// ---------------------------------------------------------------------------
__global__ __launch_bounds__(256) void cvtw_kernel(
    const float* __restrict__ Wq, const float* __restrict__ Wk,
    const float* __restrict__ Wv, __bf16* __restrict__ Wb) {
  const int idx = (blockIdx.x * 256 + threadIdx.x) * 4;   // < 3*131072
  const float* src = (idx < 131072) ? Wq : ((idx < 262144) ? Wk : Wv);
  v4f f = *(const v4f*)(src + (idx & 131071));
  v4bf o;
#pragma unroll
  for (int i = 0; i < 4; ++i) o[i] = (__bf16)f[i];
  *(v4bf*)(Wb + idx) = o;
}

// ---------------------------------------------------------------------------
// QKV projection: one wave computes a 16x128 tile of Q, K, or V.
// Ping-pong software pipeline (k-loop unrolled x2): while one buffer's 8
// WMMAs run, the other buffer's 20 loads are in flight -- no rotation copies.
//   grid = 512 M-tiles * 3 projections, block = 32 (one wave).
// ---------------------------------------------------------------------------
__device__ __forceinline__ void proj_load_stage(
    const float* __restrict__ xrow, const __bf16* __restrict__ W,
    int m, int kt, int kba, int kbb, v16bf& a, v16bf b[8]) {
  const int ka = kt * 32 + kba;
  const int kb = kt * 32 + kbb;
  a = cvt16bf(xrow + ka, xrow + ka + 8);
#pragma unroll
  for (int nt = 0; nt < 8; ++nt) {
    const __bf16* wp = W + (size_t)(nt * 16 + m) * DM + kb;
    b[nt] = load16bf(wp, wp + 8);
  }
}

__global__ __launch_bounds__(32) void qkv_proj_kernel(
    const float* __restrict__ X, const __bf16* __restrict__ Wb,
    __bf16* __restrict__ Qb, __bf16* __restrict__ Kb, __bf16* __restrict__ Vt) {
  const int lane = threadIdx.x & 31;
  const int m    = lane & 15;
  const int half = lane >> 4;
  const int proj = blockIdx.x >> 9;   // 0=Q 1=K 2=V
  const int mt   = blockIdx.x & 511;

  const __bf16* __restrict__ W = Wb + (size_t)proj * (DK * DM);
  const int row = mt * 16 + m;
  const float* __restrict__ xrow = X + (size_t)row * DM;
  const int kba = half * 16;   // A k-offset within 32-chunk
  const int kbb = half * 16;   // B k-offset within 32-chunk

  v8f acc[8];
#pragma unroll
  for (int nt = 0; nt < 8; ++nt) acc[nt] = (v8f){};

  v16bf a0, a1, b0[8], b1[8];
  proj_load_stage(xrow, W, m, 0, kba, kbb, a0, b0);   // kt = 0

  for (int kt = 0; kt < DM / 32; kt += 2) {
    // prefetch kt+1 into buf1 (kt <= 30, so kt+1 always valid)
    proj_load_stage(xrow, W, m, kt + 1, kba, kbb, a1, b1);
    // consume buf0 (kt)
#pragma unroll
    for (int nt = 0; nt < 8; ++nt)
      acc[nt] = wmma_bf16(a0, b0[nt], acc[nt]);
    // prefetch kt+2 into buf0 (dummy reload on last pass; harmless)
    const int nx = (kt + 2 < DM / 32) ? kt + 2 : kt;
    proj_load_stage(xrow, W, m, nx, kba, kbb, a0, b0);
    // consume buf1 (kt+1)
#pragma unroll
    for (int nt = 0; nt < 8; ++nt)
      acc[nt] = wmma_bf16(a1, b1[nt], acc[nt]);
  }

  // Uniform strided store: Q/K row-major, V transposed -> no per-element branch
  __bf16* outp;
  size_t rstride, cstride;
  if (proj == 0)      { outp = Qb; rstride = DK; cstride = 1;   }
  else if (proj == 1) { outp = Kb; rstride = DK; cstride = 1;   }
  else                { outp = Vt; rstride = 1;  cstride = SEQ; }

  // C layout: VGPR r -> row r (lanes 0-15) / r+8 (lanes 16-31), col = lane%16
#pragma unroll
  for (int nt = 0; nt < 8; ++nt) {
#pragma unroll
    for (int r = 0; r < 8; ++r) {
      const size_t orow = mt * 16 + half * 8 + r;
      const size_t ocol = nt * 16 + m;
      outp[orow * rstride + ocol * cstride] = (__bf16)acc[nt][r];
    }
  }
}

// ---------------------------------------------------------------------------
// Causal flash attention. One 16-row query tile per block; 4 waves stride the
// key range (32 keys/step, stride 128) with private flash state, then a
// log-sum-exp combine through LDS merges partials.
//   grid = 512, block = 128 (4 waves).
// ---------------------------------------------------------------------------
__global__ __launch_bounds__(128) void attn_kernel(
    const __bf16* __restrict__ Qb, const __bf16* __restrict__ Kb,
    const __bf16* __restrict__ Vt, float* __restrict__ Out) {
  __shared__ alignas(32) float  O_ws[WPB][16][DK];   // 32 KB
  __shared__ float m_ws[WPB][16];
  __shared__ float l_ws[WPB][16];
  __shared__ alignas(32) __bf16 ptile[WPB][16 * 32]; // 4 KB

  const int w     = threadIdx.x >> 5;
  const int lane  = threadIdx.x & 31;
  const int m     = lane & 15;
  const int half  = lane >> 4;
  const int qbase = blockIdx.x * 16;

  // scores * (1/sqrt(128)) * log2(e), so softmax uses exp2
  const float KSC = 0.08838834764831845f * 1.4426950408889634f;

  // Q fragments for all 4 d-chunks of 32 (same rows for every wave)
  v16bf aq[4];
  const __bf16* qrow = Qb + (size_t)(qbase + m) * DK;
#pragma unroll
  for (int c = 0; c < 4; ++c) {
    const int kb = c * 32 + half * 16;
    aq[c] = load16bf(qrow + kb, qrow + kb + 8);
  }

  float mr[8], lr[8];
  v8f o[8];
#pragma unroll
  for (int r = 0; r < 8; ++r) { mr[r] = -1e30f; lr[r] = 0.0f; }
#pragma unroll
  for (int nt = 0; nt < 8; ++nt) o[nt] = (v8f){};

  for (int jbase = w * 32; jbase < qbase + 16; jbase += 32 * WPB) {
    const int  jb0 = jbase;
    const int  jb1 = jbase + 16;
    const bool t1v = (jb1 < qbase + 16);

    // ---- issue ALL K and V fragment loads up front (32 b128s in flight) ----
    v16bf bk[2][4];
    {
      const __bf16* kr0 = Kb + (size_t)(jb0 + m) * DK + half * 16;
      const __bf16* kr1 = Kb + (size_t)((t1v ? jb1 : jb0) + m) * DK + half * 16;
#pragma unroll
      for (int c = 0; c < 4; ++c) {
        bk[0][c] = load16bf(kr0 + c * 32, kr0 + c * 32 + 8);
        bk[1][c] = load16bf(kr1 + c * 32, kr1 + c * 32 + 8);
      }
    }
    v16bf bv[8];
#pragma unroll
    for (int vt = 0; vt < 8; ++vt) {
      const __bf16* vp = Vt + (size_t)(vt * 16 + m) * SEQ + jbase + half * 16;
      bv[vt] = load16bf(vp, vp + 8);
    }

    // ---- scores + causal mask ---------------------------------------------
    float s[2][8];
#pragma unroll
    for (int t = 0; t < 2; ++t) {
      const int jb = jbase + t * 16;
      v8f sc = (v8f){};
#pragma unroll
      for (int c = 0; c < 4; ++c)
        sc = wmma_bf16(aq[c], bk[t][c], sc);
      const bool needmask = (jb + 16 > qbase);
      const bool dead     = (t == 1) && !t1v;
#pragma unroll
      for (int r = 0; r < 8; ++r) {
        float v = sc[r] * KSC;
        const int row = qbase + half * 8 + r;
        const int col = jb + m;
        if (dead || (needmask && col > row)) v = -1e30f;
        s[t][r] = v;
      }
    }

    // ---- online softmax (row = VGPR index + half*8, cols across 16 lanes) --
    float mx[8], al[8], p0[8], p1[8], rs[8];
#pragma unroll
    for (int r = 0; r < 8; ++r) mx[r] = fmaxf(s[0][r], s[1][r]);
#pragma unroll
    for (int off = 8; off >= 1; off >>= 1)
#pragma unroll
      for (int r = 0; r < 8; ++r)
        mx[r] = fmaxf(mx[r], __shfl_xor(mx[r], off, 32));
#pragma unroll
    for (int r = 0; r < 8; ++r) {
      const float mn = fmaxf(mr[r], mx[r]);
      al[r] = exp2f(mr[r] - mn);
      p0[r] = exp2f(s[0][r] - mn);
      p1[r] = exp2f(s[1][r] - mn);
      mr[r] = mn;
      rs[r] = p0[r] + p1[r];
    }
#pragma unroll
    for (int off = 8; off >= 1; off >>= 1)
#pragma unroll
      for (int r = 0; r < 8; ++r)
        rs[r] += __shfl_xor(rs[r], off, 32);
#pragma unroll
    for (int r = 0; r < 8; ++r) lr[r] = lr[r] * al[r] + rs[r];
#pragma unroll
    for (int nt = 0; nt < 8; ++nt)
#pragma unroll
      for (int r = 0; r < 8; ++r) o[nt][r] *= al[r];

    // ---- repack P (score layout) -> A-fragment layout through LDS ---------
    __bf16* pt = ptile[w];
#pragma unroll
    for (int r = 0; r < 8; ++r) {
      pt[(half * 8 + r) * 32 + m]      = (__bf16)p0[r];
      pt[(half * 8 + r) * 32 + 16 + m] = (__bf16)p1[r];
    }
    __threadfence_block();
    v16bf pa = load16bf(&pt[m * 32 + half * 16], &pt[m * 32 + half * 16 + 8]);
    __threadfence_block();

    // ---- PV: B[k][n] = V[jbase+k][vt*16+n] = Vt[col][jbase+k] --------------
#pragma unroll
    for (int vt = 0; vt < 8; ++vt)
      o[vt] = wmma_bf16(pa, bv[vt], o[vt]);
  }

  // ---- publish per-wave partials --------------------------------------------
#pragma unroll
  for (int r = 0; r < 8; ++r) {
    if (m == 0) {
      m_ws[w][half * 8 + r] = mr[r];
      l_ws[w][half * 8 + r] = lr[r];
    }
  }
#pragma unroll
  for (int nt = 0; nt < 8; ++nt)
#pragma unroll
    for (int r = 0; r < 8; ++r)
      O_ws[w][half * 8 + r][nt * 16 + m] = o[nt][r];
  __syncthreads();

  // ---- log-sum-exp combine: wave w merges column tiles 2w, 2w+1 ------------
#pragma unroll
  for (int nn = 0; nn < 2; ++nn) {
    const int nt  = w * 2 + nn;
    const int col = nt * 16 + m;
#pragma unroll
    for (int r = 0; r < 8; ++r) {
      const int row = half * 8 + r;
      float mp0 = m_ws[0][row], mp1 = m_ws[1][row];
      float mp2 = m_ws[2][row], mp3 = m_ws[3][row];
      float mt = fmaxf(fmaxf(mp0, mp1), fmaxf(mp2, mp3));
      float s0 = exp2f(mp0 - mt), s1 = exp2f(mp1 - mt);
      float s2 = exp2f(mp2 - mt), s3 = exp2f(mp3 - mt);
      float den = s0 * l_ws[0][row] + s1 * l_ws[1][row] +
                  s2 * l_ws[2][row] + s3 * l_ws[3][row];
      float num = s0 * O_ws[0][row][col] + s1 * O_ws[1][row][col] +
                  s2 * O_ws[2][row][col] + s3 * O_ws[3][row][col];
      Out[(size_t)(qbase + row) * DK + col] = num / den;
    }
  }
}

extern "C" void kernel_launch(void* const* d_in, const int* in_sizes, int n_in,
                              void* d_out, int out_size, void* d_ws, size_t ws_size,
                              hipStream_t stream) {
  (void)in_sizes; (void)n_in; (void)out_size; (void)ws_size;
  const float* X  = (const float*)d_in[0];
  const float* Wq = (const float*)d_in[1];
  const float* Wk = (const float*)d_in[2];
  const float* Wv = (const float*)d_in[3];

  __bf16* Qb = (__bf16*)d_ws;                       // SEQ x DK
  __bf16* Kb = Qb + (size_t)SEQ * DK;               // SEQ x DK
  __bf16* Vt = Kb + (size_t)SEQ * DK;               // DK x SEQ (transposed)
  __bf16* Wb = Vt + (size_t)SEQ * DK;               // 3 x (DK x DM) bf16
  float*  Out = (float*)d_out;

  hipLaunchKernelGGL(cvtw_kernel, dim3(384), dim3(256), 0, stream,
                     Wq, Wk, Wv, Wb);
  hipLaunchKernelGGL(qkv_proj_kernel, dim3(512 * 3), dim3(32), 0, stream,
                     X, Wb, Qb, Kb, Vt);
  hipLaunchKernelGGL(attn_kernel, dim3(512), dim3(WPB * 32), 0, stream,
                     Qb, Kb, Vt, Out);
}
